// CausalSelfAttention_48223892799905
// MI455X (gfx1250) — matrix operand order repaired and checked by reading, not compiled, not run on hardware
//
#include <hip/hip_runtime.h>
#include <hip/hip_bf16.h>
#include <stdint.h>

// ---------------------------------------------------------------------------
// CausalSelfAttention forward for MI455X (gfx1250, wave32, WMMA + TDM).
// B=4, T=2048, C=2048, H=16, HD=128.
// Heavy math: v_wmma_f32_16x16x32_bf16. Tile staging: tensor_load_to_lds (TDM)
// with double buffering (s_wait_tensorcnt 1) so DMA overlaps the WMMA stream.
// ---------------------------------------------------------------------------

typedef __attribute__((ext_vector_type(16))) __bf16 v16bf;
typedef __attribute__((ext_vector_type(8)))  float  v8f;
typedef unsigned int u32x4 __attribute__((ext_vector_type(4)));
typedef int          i32x4 __attribute__((ext_vector_type(4)));
typedef int          i32x8 __attribute__((ext_vector_type(8)));

#define CSA_B  4
#define CSA_T  2048
#define CSA_C  2048
#define CSA_H  16
#define CSA_HD 128
#define CSA_M  (CSA_B * CSA_T)      // 8192
#define CSA_3C (3 * CSA_C)          // 6144

#if defined(__has_builtin)
#  if __has_builtin(__builtin_amdgcn_tensor_load_to_lds)
#    define CSA_HAS_TDM 1
#  endif
#endif
#if __has_include(<hip/amd_detail/amd_gfx1250_TDM.h>)
#  define CSA_TDM6 1   // therock/clang-23 6-arg form
#endif

__device__ __forceinline__ v8f wmma_bf16(const v16bf& a, const v16bf& b, const v8f& c) {
  return __builtin_amdgcn_wmma_f32_16x16x32_bf16(
      false, a, false, b, (short)0, c, false, false);
}

#if defined(CSA_HAS_TDM)
// One TDM 2-D tile load: rows x cols bf16, global row stride `stride` (elems),
// into LDS at byte offset `lds_addr`, with LDS padding of `padAmtDW` dwords
// every (2 << padIntCode) dwords.
__device__ __forceinline__ void csa_tdm_load_2d(unsigned lds_addr,
                                                const __bf16* gptr,
                                                int cols, int rows, int stride,
                                                int padIntCode, int padAmtDW) {
  unsigned long long ga = (unsigned long long)(const void*)gptr;
  u32x4 g0;
  g0.x = 1u;                                   // count=1, user descriptor
  g0.y = lds_addr;
  g0.z = (unsigned)ga;                         // global_addr[31:0]
  g0.w = (unsigned)(ga >> 32) | (2u << 30);    // global_addr[56:32] | type=2
  int padEn  = padAmtDW > 0 ? 1 : 0;
  int padAmt = padAmtDW > 0 ? (padAmtDW - 1) : 0;
  i32x8 g1;
  g1[0] = (1 << 16)                            // data_size = 2 bytes
        | (padEn << 20) | (padIntCode << 22) | (padAmt << 25);
  g1[1] = (int)(((unsigned)cols & 0xFFFFu) << 16);            // tensor_dim0 lo
  g1[2] = (int)((((unsigned)cols >> 16) & 0xFFFFu)
        | (((unsigned)rows & 0xFFFFu) << 16));                // dim0 hi | dim1 lo
  g1[3] = (int)((((unsigned)rows >> 16) & 0xFFFFu)
        | (((unsigned)cols & 0xFFFFu) << 16));                // dim1 hi | tile_dim0
  g1[4] = rows & 0xFFFF;                                      // tile_dim1
  g1[5] = stride;                                             // dim0_stride lo
  g1[6] = 0;
  g1[7] = 0;
  i32x4 z4 = (i32x4){0, 0, 0, 0};
#if defined(CSA_TDM6)
  i32x8 z8 = (i32x8){0, 0, 0, 0, 0, 0, 0, 0};
  __builtin_amdgcn_tensor_load_to_lds(g0, g1, z4, z4, z8, 0);
#else
  __builtin_amdgcn_tensor_load_to_lds(g0, g1, z4, z4, 0);
#endif
}

// SIMM16 of s_wait_tensorcnt must be a literal: branch on the uniform flag.
__device__ __forceinline__ void csa_wait_tensor(bool oneOutstanding) {
  if (oneOutstanding) __builtin_amdgcn_s_wait_tensorcnt(1);
  else                __builtin_amdgcn_s_wait_tensorcnt(0);
}
#endif

// ---------------------------------------------------------------------------
// Elementwise f32 -> bf16 conversion (n divisible by 1024), NT loads.
// ---------------------------------------------------------------------------
__global__ __launch_bounds__(256)
void csa_cvt_bf16(const float* __restrict__ in, __bf16* __restrict__ out,
                  long long n) {
  long long i = ((long long)blockIdx.x * blockDim.x + threadIdx.x) * 4;
#pragma unroll
  for (int j = 0; j < 4; ++j)
    out[i + j] = (__bf16)__builtin_nontemporal_load(&in[i + j]);
}

// ---------------------------------------------------------------------------
// GEMM: Cm[M,N] (f32) = A[M,K](bf16) * Bm[K,N](bf16) + bias[N]
// 256 thr = 8 waves, tile 128x128, K-step 32, DOUBLE-BUFFERED:
// A-tile via TDM (padded LDS rows), B-tile transposed via global_b128+scatter.
// Tile k+1 is issued before waiting on tile k (s_wait_tensorcnt 1).
// ---------------------------------------------------------------------------
__global__ __launch_bounds__(256)
void csa_gemm(const __bf16* __restrict__ A, const __bf16* __restrict__ Bm,
              const float* __restrict__ bias, float* __restrict__ Cm,
              int M, int N, int K) {
  __shared__ __bf16 As[2][128][40];    // 80 B rows (32 data + 8 pad), 16B-aligned
  __shared__ __bf16 BsT[2][128][40];   // transposed: [col][k], padded

  const int tid  = threadIdx.x;
  const int lane = tid & 31;
  const int wave = tid >> 5;
  const int wm   = wave >> 2;          // 0..1
  const int wn   = wave & 3;           // 0..3
  const int half = lane >> 4;
  const int l16  = lane & 15;

  const int bm = blockIdx.x * 128;
  const int bn = blockIdx.y * 128;

  v8f acc[4][2];
#pragma unroll
  for (int sm = 0; sm < 4; ++sm)
#pragma unroll
    for (int sn = 0; sn < 2; ++sn)
      acc[sm][sn] = (v8f){0.f, 0.f, 0.f, 0.f, 0.f, 0.f, 0.f, 0.f};

  const int nSteps = K >> 5;

  // ---- prologue: stage tile 0 into buffer 0.
#pragma unroll
  for (int i = 0; i < 16; ++i) {
    int e = tid * 16 + i;
    int r = e >> 7, c = e & 127;
    BsT[0][c][r] = Bm[(size_t)r * N + (bn + c)];
  }
#if defined(CSA_HAS_TDM)
  if (wave == 0)
    csa_tdm_load_2d((unsigned)(unsigned long long)&As[0][0][0],
                    A + (size_t)bm * K, 32, 128, K, /*padInt=*/3, /*padDW=*/4);
#else
#pragma unroll
  for (int i = 0; i < 16; ++i) {
    int e = tid * 16 + i;
    int r = e >> 5, c = e & 31;
    As[0][r][c] = A[(size_t)(bm + r) * K + c];
  }
#endif

  int buf = 0;
  for (int kt = 0; kt < nSteps; ++kt) {
    const int nxt  = buf ^ 1;
    const bool more = (kt + 1) < nSteps;

    // ---- prefetch tile kt+1 into the other buffer (overlaps with wait).
    if (more) {
      const int k1 = (kt + 1) * 32;
#pragma unroll
      for (int i = 0; i < 16; ++i) {
        int e = tid * 16 + i;
        int r = e >> 7, c = e & 127;
        BsT[nxt][c][r] = Bm[(size_t)(k1 + r) * N + (bn + c)];
      }
#if defined(CSA_HAS_TDM)
      if (wave == 0)
        csa_tdm_load_2d((unsigned)(unsigned long long)&As[nxt][0][0],
                        A + (size_t)bm * K + k1, 32, 128, K, 3, 4);
#else
#pragma unroll
      for (int i = 0; i < 16; ++i) {
        int e = tid * 16 + i;
        int r = e >> 5, c = e & 31;
        As[nxt][r][c] = A[(size_t)(bm + r) * K + (k1 + c)];
      }
#endif
      if (k1 + 32 < K)  // gfx1250 global_prefetch_b8 two tiles ahead
        __builtin_prefetch(&Bm[(size_t)(k1 + 32 + (tid >> 4)) * N + bn + (tid & 15) * 8], 0, 1);
    }
#if defined(CSA_HAS_TDM)
    if (wave == 0) csa_wait_tensor(more);
#endif
    __syncthreads();   // tile kt ready & visible

    // ---- fragments: all gathers contiguous -> ds_load_b128 pairs.
    v16bf af[4], bfr[2];
#pragma unroll
    for (int sm = 0; sm < 4; ++sm) {
      int row = wm * 64 + sm * 16 + l16;
#pragma unroll
      for (int e = 0; e < 16; ++e) {
        int kk = half * 16 + e;
        af[sm][e] = As[buf][row][kk];
      }
    }
#pragma unroll
    for (int sn = 0; sn < 2; ++sn) {
      int col = wn * 32 + sn * 16 + l16;
#pragma unroll
      for (int e = 0; e < 16; ++e)
        bfr[sn][e] = BsT[buf][col][half * 16 + e];
    }
#pragma unroll
    for (int sm = 0; sm < 4; ++sm)
#pragma unroll
      for (int sn = 0; sn < 2; ++sn)
        acc[sm][sn] = wmma_bf16(af[sm], bfr[sn], acc[sm][sn]);
    __syncthreads();   // compute done before buffer `buf` is overwritten
    buf = nxt;
  }

  // ---- epilogue: streaming f32 stores (qkv > L2, use NT policy).
#pragma unroll
  for (int sm = 0; sm < 4; ++sm)
#pragma unroll
    for (int sn = 0; sn < 2; ++sn)
#pragma unroll
      for (int j = 0; j < 8; ++j) {
        int row = bm + wm * 64 + sm * 16 + (half ? j + 8 : j);
        int col = bn + wn * 32 + sn * 16 + l16;
        __builtin_nontemporal_store(acc[sm][sn][j] + bias[col],
                                    &Cm[(size_t)row * N + col]);
      }
}

// ---------------------------------------------------------------------------
// RoPE + repack: qkv[B,T,3C] (f32) -> qb,kb,vb[B,H,T,HD] (bf16), q pre-scaled.
// ---------------------------------------------------------------------------
__global__ __launch_bounds__(256)
void csa_rope_pack(const float* __restrict__ qkv,
                   const float* __restrict__ cosb, const float* __restrict__ sinb,
                   __bf16* __restrict__ qb, __bf16* __restrict__ kb,
                   __bf16* __restrict__ vb) {
  long long p = (long long)blockIdx.x * blockDim.x + threadIdx.x;
  int i = (int)(p & 63);
  long long r = p >> 6;
  int h = (int)(r & (CSA_H - 1));
  r >>= 4;
  int t = (int)(r & (CSA_T - 1));
  int b = (int)(r >> 11);

  size_t src = ((size_t)(b * CSA_T + t)) * CSA_3C + h * CSA_HD + 2 * i;
  size_t dst = (((size_t)(b * CSA_H + h)) * CSA_T + t) * CSA_HD + 2 * i;
  float c = cosb[t * 64 + i];
  float s = sinb[t * 64 + i];
  const float scale = 0.08838834764831845f;   // 1/sqrt(128), folded into q

  float qr = qkv[src], qi = qkv[src + 1];
  qb[dst]     = (__bf16)((qr * c - qi * s) * scale);
  qb[dst + 1] = (__bf16)((qr * s + qi * c) * scale);
  float kr = qkv[src + CSA_C], ki = qkv[src + CSA_C + 1];
  kb[dst]     = (__bf16)(kr * c - ki * s);
  kb[dst + 1] = (__bf16)(kr * s + ki * c);
  vb[dst]     = (__bf16)qkv[src + 2 * CSA_C];
  vb[dst + 1] = (__bf16)qkv[src + 2 * CSA_C + 1];
}

// ---------------------------------------------------------------------------
// Flash attention: block = (b,h,64-row q tile); 128 thr = 4 waves.
// K-tile by TDM, V-tile transposed manually, both DOUBLE-BUFFERED.
// QK^T and P*V on bf16 WMMA; online softmax via wave32 half-group shuffles.
// ---------------------------------------------------------------------------
__global__ __launch_bounds__(128)
void csa_flash_attn(const __bf16* __restrict__ qb, const __bf16* __restrict__ kb,
                    const __bf16* __restrict__ vb, __bf16* __restrict__ yb) {
  __shared__ __bf16 Ks[2][32][136];    // 272 B rows (128 data + 8 pad)
  __shared__ __bf16 VsT[2][128][40];   // transposed V: [hd][key], padded
  __shared__ __bf16 Ps[4][16][32];     // per-wave P scratch

  const int tid  = threadIdx.x;
  const int lane = tid & 31;
  const int wave = tid >> 5;
  const int half = lane >> 4;
  const int l16  = lane & 15;

  const int nQB = CSA_T / 64;
  int blk = blockIdx.x;
  int qb_i = blk % nQB;
  int bh   = blk / nQB;
  int h    = bh % CSA_H;
  int b    = bh / CSA_H;
  int q0   = qb_i * 64;

  const size_t headBase = ((size_t)(b * CSA_H + h)) * CSA_T * CSA_HD;

  // Q stripe (16x128) as 4 A-fragments (pre-scaled in qb).
  v16bf qf[4];
  {
    const __bf16* qptr = qb + headBase + (size_t)(q0 + wave * 16 + l16) * CSA_HD;
#pragma unroll
    for (int c4 = 0; c4 < 4; ++c4)
#pragma unroll
      for (int e = 0; e < 16; ++e) {
        int kk = c4 * 32 + half * 16 + e;
        qf[c4][e] = qptr[kk];
      }
  }

  v8f acc[8];
#pragma unroll
  for (int nt = 0; nt < 8; ++nt)
    acc[nt] = (v8f){0.f, 0.f, 0.f, 0.f, 0.f, 0.f, 0.f, 0.f};
  float m[8], l[8];
#pragma unroll
  for (int j = 0; j < 8; ++j) { m[j] = -3.0e38f; l[j] = 0.f; }

  const int nk = (q0 + 64) / 32;

  // ---- prologue: stage key-tile 0 into buffer 0.
#pragma unroll
  for (int i = 0; i < 32; ++i) {
    int e = tid * 32 + i;
    int r = e >> 7, c = e & 127;
    VsT[0][c][r] = vb[headBase + (size_t)r * CSA_HD + c];
  }
#if defined(CSA_HAS_TDM)
  if (wave == 0)
    csa_tdm_load_2d((unsigned)(unsigned long long)&Ks[0][0][0],
                    kb + headBase, 128, 32, CSA_HD, /*padInt=*/5, /*padDW=*/4);
#else
#pragma unroll
  for (int i = 0; i < 32; ++i) {
    int e = tid * 32 + i;
    int r = e >> 7, c = e & 127;
    Ks[0][r][c] = kb[headBase + (size_t)r * CSA_HD + c];
  }
#endif

  int buf = 0;
  for (int kt = 0; kt < nk; ++kt) {
    const int kbase = kt * 32;
    const int nxt   = buf ^ 1;
    const bool more = (kt + 1) < nk;

    // ---- prefetch key-tile kt+1.
    if (more) {
      const int k1 = kbase + 32;
#pragma unroll
      for (int i = 0; i < 32; ++i) {
        int e = tid * 32 + i;
        int r = e >> 7, c = e & 127;
        VsT[nxt][c][r] = vb[headBase + (size_t)(k1 + r) * CSA_HD + c];
      }
#if defined(CSA_HAS_TDM)
      if (wave == 0)
        csa_tdm_load_2d((unsigned)(unsigned long long)&Ks[nxt][0][0],
                        kb + headBase + (size_t)k1 * CSA_HD, 128, 32, CSA_HD, 5, 4);
#else
#pragma unroll
      for (int i = 0; i < 32; ++i) {
        int e = tid * 32 + i;
        int r = e >> 7, c = e & 127;
        Ks[nxt][r][c] = kb[headBase + (size_t)(k1 + r) * CSA_HD + c];
      }
#endif
    }
#if defined(CSA_HAS_TDM)
    if (wave == 0) csa_wait_tensor(more);
#endif
    __syncthreads();   // tile kt ready & visible

    // ---- S = Q K^T : two 16x16 score tiles.
    v8f s[2];
#pragma unroll
    for (int n = 0; n < 2; ++n) {
      s[n] = (v8f){0.f, 0.f, 0.f, 0.f, 0.f, 0.f, 0.f, 0.f};
#pragma unroll
      for (int c4 = 0; c4 < 4; ++c4) {
        v16bf kf;
#pragma unroll
        for (int e = 0; e < 16; ++e)
          kf[e] = Ks[buf][n * 16 + l16][c4 * 32 + half * 16 + e];
        s[n] = wmma_bf16(qf[c4], kf, s[n]);
      }
    }

    // ---- causal mask + online softmax.
    float rs[8];
#pragma unroll
    for (int j = 0; j < 8; ++j) {
      int gr = q0 + wave * 16 + (half ? j + 8 : j);
      float s0 = (kbase + l16      > gr) ? -3.0e38f : s[0][j];
      float s1 = (kbase + 16 + l16 > gr) ? -3.0e38f : s[1][j];
      float mx = fmaxf(s0, s1);
#pragma unroll
      for (int off = 1; off < 16; off <<= 1)
        mx = fmaxf(mx, __shfl_xor(mx, off, 32));
      float nm   = fmaxf(m[j], mx);
      float corr = __expf(m[j] - nm);
      float p0   = __expf(s0 - nm);
      float p1   = __expf(s1 - nm);
      float sum  = p0 + p1;
#pragma unroll
      for (int off = 1; off < 16; off <<= 1)
        sum += __shfl_xor(sum, off, 32);
      l[j] = l[j] * corr + sum;
      m[j] = nm;
      rs[j] = corr;
      int pr = half ? j + 8 : j;
      Ps[wave][pr][l16]      = (__bf16)p0;
      Ps[wave][pr][16 + l16] = (__bf16)p1;
    }
    asm volatile("" ::: "memory");   // keep intra-wave P stores before loads

#pragma unroll
    for (int nt = 0; nt < 8; ++nt)
#pragma unroll
      for (int j = 0; j < 8; ++j)
        acc[nt][j] *= rs[j];

    v16bf pf;
#pragma unroll
    for (int e = 0; e < 16; ++e) {
      int kk = half * 16 + e;
      pf[e] = Ps[wave][l16][kk];
    }
#pragma unroll
    for (int nt = 0; nt < 8; ++nt) {
      v16bf vf;
#pragma unroll
      for (int e = 0; e < 16; ++e)
        vf[e] = VsT[buf][nt * 16 + l16][half * 16 + e];
      acc[nt] = wmma_bf16(pf, vf, acc[nt]);
    }
    __syncthreads();   // compute done before buffer `buf` is overwritten
    buf = nxt;
  }

  // ---- normalize, write yb in [B,T,C] layout (bf16) for the proj GEMM.
#pragma unroll
  for (int nt = 0; nt < 8; ++nt)
#pragma unroll
    for (int j = 0; j < 8; ++j) {
      int row = q0 + wave * 16 + (half ? j + 8 : j);
      int col = h * CSA_HD + nt * 16 + l16;
      yb[(size_t)(b * CSA_T + row) * CSA_C + col] = (__bf16)(acc[nt][j] / l[j]);
    }
}

// ---------------------------------------------------------------------------
extern "C" void kernel_launch(void* const* d_in, const int* in_sizes, int n_in,
                              void* d_out, int out_size, void* d_ws, size_t ws_size,
                              hipStream_t stream) {
  const float* x     = (const float*)d_in[0];
  const float* fcos  = (const float*)d_in[1];
  const float* fsin  = (const float*)d_in[2];
  const float* Wqkv  = (const float*)d_in[3];
  const float* bqkv  = (const float*)d_in[4];
  const float* Wproj = (const float*)d_in[5];
  const float* bproj = (const float*)d_in[6];
  float* out = (float*)d_out;

  // Workspace carve-up.
  char* ws = (char*)d_ws;
  float*  qkv    = (float*)ws;            ws += (size_t)CSA_M * CSA_3C * 4;   // 201.3 MB
  __bf16* xb     = (__bf16*)ws;           ws += (size_t)CSA_M * CSA_C * 2;    //  33.6 MB
  __bf16* Wqkvb  = (__bf16*)ws;           ws += (size_t)CSA_C * CSA_3C * 2;   //  25.2 MB
  __bf16* Wprojb = (__bf16*)ws;           ws += (size_t)CSA_C * CSA_C * 2;    //   8.4 MB
  __bf16* qbb    = (__bf16*)ws;           ws += (size_t)CSA_M * CSA_C * 2;
  __bf16* kbb    = (__bf16*)ws;           ws += (size_t)CSA_M * CSA_C * 2;
  __bf16* vbb    = (__bf16*)ws;           ws += (size_t)CSA_M * CSA_C * 2;
  __bf16* ybb    = (__bf16*)ws;           ws += (size_t)CSA_M * CSA_C * 2;

  // 0) one-time bf16 conversions
  csa_cvt_bf16<<<(int)(((long long)CSA_M * CSA_C) / 1024), 256, 0, stream>>>(
      x, xb, (long long)CSA_M * CSA_C);
  csa_cvt_bf16<<<(int)(((long long)CSA_C * CSA_3C) / 1024), 256, 0, stream>>>(
      Wqkv, Wqkvb, (long long)CSA_C * CSA_3C);
  csa_cvt_bf16<<<(int)(((long long)CSA_C * CSA_C) / 1024), 256, 0, stream>>>(
      Wproj, Wprojb, (long long)CSA_C * CSA_C);

  // 1) qkv = x @ Wqkv + bqkv
  {
    dim3 grid(CSA_M / 128, CSA_3C / 128);    // 64 x 48
    csa_gemm<<<grid, 256, 0, stream>>>(xb, Wqkvb, bqkv, qkv,
                                       CSA_M, CSA_3C, CSA_C);
  }
  // 2) RoPE + head-major bf16 repack
  {
    long long pairs = (long long)CSA_M * CSA_H * 64;   // 8,388,608
    csa_rope_pack<<<(int)(pairs / 256), 256, 0, stream>>>(qkv, fcos, fsin,
                                                          qbb, kbb, vbb);
  }
  // 3) flash attention -> yb (bf16)
  {
    int blocks = CSA_B * CSA_H * (CSA_T / 64);         // 2048
    csa_flash_attn<<<blocks, 128, 0, stream>>>(qbb, kbb, vbb, ybb);
  }
  // 4) out = y @ Wproj + bproj
  {
    dim3 grid(CSA_M / 128, CSA_C / 128);               // 64 x 16
    csa_gemm<<<grid, 256, 0, stream>>>(ybb, Wprojb, bproj, out,
                                       CSA_M, CSA_C, CSA_C);
  }
}
